// STB_77171972374880
// MI455X (gfx1250) — matrix operand order repaired and checked by reading, not compiled, not run on hardware
//
#include <hip/hip_runtime.h>
#include <hip/hip_bf16.h>
#include <math.h>

// ---------------------------------------------------------------------------
// Problem constants (from the reference)
// ---------------------------------------------------------------------------
#define Sn    8
#define Hn    128
#define Wn    128
#define Pn    (Hn * Wn)        // 16384 pixels
#define DISPn 6
#define Kn    13               // 2*DISP+1
#define CHI   64
#define CLO   128
#define CINF  192              // feature channels = CHI + CLO
#define Fc    64               // matching feature channels (OUTP)
#define Ec    64               // embedding channels
#define SEC   256              // SE channels (Ec + CINF)
#define SEH   16               // SEC / 16
#define RP    (Hn + 2 * DISPn) // 140 padded side
#define LOH   64               // feat_lo H (= Hn/2)

typedef __attribute__((ext_vector_type(16))) _Float16 v16h;
typedef __attribute__((ext_vector_type(8)))  _Float16 v8h;
typedef __attribute__((ext_vector_type(8)))  float    v8f;
typedef __attribute__((__vector_size__(16))) int      vi4;   // 16B chunk for async DMA

struct HPair { v8h lo, hi; };
union HU { HPair p; v16h v; };

// gfx1250 async global->LDS path (ASYNCcnt), with graceful fallback.
#if defined(__gfx1250__) && \
    __has_builtin(__builtin_amdgcn_global_load_async_to_lds_b128) && \
    __has_builtin(__builtin_amdgcn_s_wait_asynccnt)
#define ASYNC_OK 1
#else
#define ASYNC_OK 0
#endif

// ---------------------------------------------------------------------------
// Weight repack: OIHW fp32 -> [OCp][ky*kx][ICp] f16 (zero padded)
// ---------------------------------------------------------------------------
__global__ void k_reorder_w(const float* __restrict__ src, _Float16* __restrict__ dst,
                            int OC, int IC, int KK, int OCp, int ICp) {
  int idx = blockIdx.x * blockDim.x + threadIdx.x;
  int total = OCp * KK * KK * ICp;
  if (idx >= total) return;
  int icp = idx % ICp;
  int t   = idx / ICp;
  int tap = t % (KK * KK);
  int oc  = t / (KK * KK);
  float v = 0.0f;
  if (oc < OC && icp < IC) {
    int ky = tap / KK, kx = tap % KK;
    v = src[(((size_t)oc * IC + icp) * KK + ky) * KK + kx];
  }
  dst[idx] = (_Float16)v;
}

// ---------------------------------------------------------------------------
// Build feature = concat(feat_hi, bilinear_up(feat_lo)) directly into the
// channel-last m_input buffer at channel offset 64 (channels 64..255).
// align_corners=True bilinear, 64 -> 128.
// ---------------------------------------------------------------------------
__global__ void k_build_feature(const float* __restrict__ hi, const float* __restrict__ lo,
                                _Float16* __restrict__ minp) {
  int idx = blockIdx.x * blockDim.x + threadIdx.x;   // s*Pn + p
  if (idx >= Sn * Pn) return;
  int s = idx / Pn, p = idx % Pn;
  int x = p & (Wn - 1), y = p >> 7;
  _Float16* outp = minp + (size_t)idx * SEC + Ec;    // feature starts at ch 64

  const float* hp = hi + (size_t)s * CHI * Pn + p;
  for (int c = 0; c < CHI; ++c) outp[c] = (_Float16)hp[(size_t)c * Pn];

  const float scale = 63.0f / 127.0f;
  float fy = y * scale, fx = x * scale;
  int y0 = (int)floorf(fy); if (y0 > LOH - 1) y0 = LOH - 1; if (y0 < 0) y0 = 0;
  int x0 = (int)floorf(fx); if (x0 > LOH - 1) x0 = LOH - 1; if (x0 < 0) x0 = 0;
  int y1 = y0 + 1 < LOH ? y0 + 1 : LOH - 1;
  int x1 = x0 + 1 < LOH ? x0 + 1 : LOH - 1;
  float wy = fy - (float)y0, wx = fx - (float)x0;
  float w00 = (1.f - wy) * (1.f - wx), w01 = (1.f - wy) * wx;
  float w10 = wy * (1.f - wx),         w11 = wy * wx;
  int i00 = y0 * LOH + x0, i01 = y0 * LOH + x1;
  int i10 = y1 * LOH + x0, i11 = y1 * LOH + x1;
  const float* lb = lo + (size_t)s * CLO * (LOH * LOH);
  for (int c = 0; c < CLO; ++c) {
    const float* lc = lb + (size_t)c * (LOH * LOH);
    float v = lc[i00] * w00 + lc[i01] * w01 + lc[i10] * w10 + lc[i11] * w11;
    outp[CHI + c] = (_Float16)v;
  }
}

// ---------------------------------------------------------------------------
// Implicit-GEMM conv via WMMA f32 = f16 x f16.
// One wave computes noc*16 out-channels x 16 pixels (noc accumulators).
// The input strip (ksize rows x (16+2*pad) cols x cin halves) is staged into
// LDS once per block — via gfx1250 async global->LDS DMA when available —
// then every B fragment is a single aligned 32B LDS read, reused by `noc`
// back-to-back WMMAs (B global traffic /4 vs per-oc-tile fetch).
//
// A-frag (16-bit A 16x32): lane row = lane&15, K-half = lane>>4
//   elems 0..15 -> K = 16*half + 0..15
// B-frag (16-bit B 32x16): lane col = lane&15, K = 16*half + elem
// C/D: col = lane&15, row = 8*half + elem
// ---------------------------------------------------------------------------
__global__ __launch_bounds__(32)
void k_conv_wmma(const _Float16* __restrict__ in_cl, int cstride, int coff, int cin,
                 int ksize, const _Float16* __restrict__ wh, const float* __restrict__ bias,
                 int oc_real, int noc, float* __restrict__ out_nchw, int lrelu) {
  extern __shared__ _Float16 smem[];
  const int lane = threadIdx.x;
  const int col  = lane & 15;
  const int half = lane >> 4;
  const int p0   = blockIdx.x * 16;
  const int s    = blockIdx.z;
  const int x0   = p0 & (Wn - 1);
  const int y    = p0 >> 7;
  const int pad  = ksize >> 1;
  const int cols = 16 + 2 * pad;
  const int taps = ksize * ksize;
  const int strip = ksize * cols * cin;              // halves

  // ---- stage input strip into LDS (zero-filled outside the image) ----
  for (int i = lane * 8; i < strip; i += 32 * 8) {
    int ch = i % cin;
    int t2 = i / cin;
    int cc = t2 % cols;
    int r  = t2 / cols;
    int gy = y - pad + r;
    int gx = x0 - pad + cc;
    bool vld = ((unsigned)gy < (unsigned)Hn) && ((unsigned)gx < (unsigned)Wn);
#if ASYNC_OK
    if (vld) {
      const _Float16* gp = in_cl +
          ((size_t)(s * Pn + gy * Wn + gx)) * cstride + coff + ch;
      __builtin_amdgcn_global_load_async_to_lds_b128(
          (__attribute__((address_space(1))) vi4*)(void*)gp,
          (__attribute__((address_space(3))) vi4*)(void*)(smem + i),
          /*offset=*/0, /*cpol=*/0);
    } else {
      v8h z = {};
      *(v8h*)(smem + i) = z;
    }
#else
    v8h v = {};
    if (vld) v = *(const v8h*)(in_cl +
                  ((size_t)(s * Pn + gy * Wn + gx)) * cstride + coff + ch);
    *(v8h*)(smem + i) = v;
#endif
  }
#if ASYNC_OK
  __builtin_amdgcn_s_wait_asynccnt(0);
#endif
  __syncthreads();

  // ---- GEMM main loop: 1 B load feeds `noc` WMMAs ----
  v8f acc[4] = {{}, {}, {}, {}};
  for (int tap = 0; tap < taps; ++tap) {
    const int dy = tap / ksize;                      // 0..ksize-1
    const int dx = tap % ksize;
    const _Float16* brow = smem + ((size_t)(dy * cols + col + dx)) * cin + half * 16;
    for (int k0 = 0; k0 < cin; k0 += 32) {
      v16h b = *(const v16h*)(brow + k0);
      #pragma unroll
      for (int t = 0; t < 4; ++t) {
        if (t >= noc) break;
        const _Float16* ab = wh +
            ((size_t)(t * 16 + col) * taps + tap) * cin + half * 16 + k0;
        HU u;
        u.p.lo = *(const v8h*)ab;
        u.p.hi = *(const v8h*)(ab + 8);
        acc[t] = __builtin_amdgcn_wmma_f32_16x16x32_f16(
            /*neg_a=*/false, u.v, /*neg_b=*/false, b,
            /*c_mod=*/(short)0, acc[t], /*reuse_a=*/false, /*reuse_b=*/false);
      }
    }
  }

  // ---- epilogue: bias + LeakyReLU, write NCHW fp32 ----
  const int prow = p0 + col;
  #pragma unroll
  for (int t = 0; t < 4; ++t) {
    if (t >= noc) break;
    #pragma unroll
    for (int v = 0; v < 8; ++v) {
      int oc = t * 16 + half * 8 + v;
      if (oc < oc_real) {
        float r = acc[t][v] + bias[oc];
        if (lrelu) r = (r > 0.0f) ? r : 0.01f * r;
        out_nchw[((size_t)s * oc_real + oc) * Pn + prow] = r;
      }
    }
  }
}

// ---------------------------------------------------------------------------
// InstanceNorm over H*W per (s,c): normalize NCHW fp32 in place and emit a
// channel-last f16 copy for the next GEMM stage. grid = S*C blocks.
// ---------------------------------------------------------------------------
__global__ void k_instnorm(float* __restrict__ xn, const float* __restrict__ g,
                           const float* __restrict__ be, _Float16* __restrict__ ocl,
                           int cstride, int C) {
  int c = blockIdx.x % C;
  int s = blockIdx.x / C;
  float* xp = xn + ((size_t)s * C + c) * Pn;
  float sum = 0.f, sq = 0.f;
  for (int p = threadIdx.x; p < Pn; p += blockDim.x) {
    float v = xp[p]; sum += v; sq += v * v;
  }
  __shared__ float s1[256], s2[256];
  s1[threadIdx.x] = sum; s2[threadIdx.x] = sq;
  __syncthreads();
  for (int st = 128; st > 0; st >>= 1) {
    if (threadIdx.x < st) {
      s1[threadIdx.x] += s1[threadIdx.x + st];
      s2[threadIdx.x] += s2[threadIdx.x + st];
    }
    __syncthreads();
  }
  float mean = s1[0] * (1.0f / Pn);
  float var  = s2[0] * (1.0f / Pn) - mean * mean;
  float inv  = rsqrtf(var + 1e-5f) * g[c];
  float bb   = be[c];
  for (int p = threadIdx.x; p < Pn; p += blockDim.x) {
    float v = (xp[p] - mean) * inv + bb;
    xp[p] = v;
    ocl[((size_t)s * Pn + p) * cstride + c] = (_Float16)v;
  }
}

// ---------------------------------------------------------------------------
// Reference-frame seg-embedding input: cat(matching_input[0], fg, bg), padded
// from 66 to 96 channels with zeros (keeps GEMM K a multiple of 32).
// ---------------------------------------------------------------------------
__global__ void k_build_ref(const _Float16* __restrict__ micl, const float* __restrict__ seg,
                            _Float16* __restrict__ refcl) {
  int p = blockIdx.x * blockDim.x + threadIdx.x;
  if (p >= Pn) return;
  _Float16* o = refcl + (size_t)p * 96;
  const _Float16* m = micl + (size_t)p * Fc;          // s == 0
  for (int c = 0; c < Fc; ++c) o[c] = m[c];
  float l0 = seg[p];                                  // seg_logits[0,0,p]
  float l1 = seg[Pn + p];                             // seg_logits[0,1,p]
  float lab = (l1 > l0) ? 1.0f : 0.0f;
  o[64] = (_Float16)lab;
  o[65] = (_Float16)(1.0f - lab);
  for (int c = 66; c < 96; ++c) o[c] = (_Float16)0.0f;
}

// ---------------------------------------------------------------------------
// Zero-pad r (1,64,H,W channel-last f16) into [RP*RP][64].
// ---------------------------------------------------------------------------
__global__ void k_pad_r(const _Float16* __restrict__ rcl, _Float16* __restrict__ rpad) {
  int idx = blockIdx.x * blockDim.x + threadIdx.x;    // yy*RP + xx
  if (idx >= RP * RP) return;
  int xx = idx % RP, yy = idx / RP;
  _Float16* o = rpad + (size_t)idx * Ec;
  int iy = yy - DISPn, ix = xx - DISPn;
  if ((unsigned)iy < (unsigned)Hn && (unsigned)ix < (unsigned)Wn) {
    const _Float16* src = rcl + ((size_t)(iy * Wn + ix)) * Ec;
    for (int c = 0; c < Ec; ++c) o[c] = src[c];
  } else {
    for (int c = 0; c < Ec; ++c) o[c] = (_Float16)0.0f;
  }
}

// ---------------------------------------------------------------------------
// Windowed correlation (13x13, E=64) + argmax + gather of the matched
// reference embedding into m_input channels 0..63, plus coor_info ints.
// q held in registers (4x v16h); r_pad (~2.5MB) is L2-resident and shared
// across all 8 frames.
// ---------------------------------------------------------------------------
__global__ void k_corr(const _Float16* __restrict__ qcl, const _Float16* __restrict__ rpad,
                       _Float16* __restrict__ minp, int* __restrict__ coor) {
  int idx = blockIdx.x * blockDim.x + threadIdx.x;    // s*Pn + p
  if (idx >= Sn * Pn) return;
  int s = idx / Pn, p = idx % Pn;
  int x = p & (Wn - 1), y = p >> 7;

  const v16h* qp = (const v16h*)(qcl + (size_t)idx * Ec);
  v16h q0 = qp[0], q1 = qp[1], q2 = qp[2], q3 = qp[3];

  float best = -3.4e38f;
  int   bi   = 0;
  for (int i = 0; i < Kn * Kn; ++i) {
    int dy = i / Kn, dx = i % Kn;
    const _Float16* rb = rpad + ((size_t)((y + dy) * RP + (x + dx))) * Ec;
    __builtin_prefetch(rb + RP * Ec, 0, 1);           // next window row
    const v16h* rp = (const v16h*)rb;
    v16h r0 = rp[0], r1 = rp[1], r2 = rp[2], r3 = rp[3];
    float d = 0.0f;
    #pragma unroll
    for (int e = 0; e < 16; ++e) {
      d += (float)q0[e] * (float)r0[e];
      d += (float)q1[e] * (float)r1[e];
      d += (float)q2[e] * (float)r2[e];
      d += (float)q3[e] * (float)r3[e];
    }
    if (d > best) { best = d; bi = i; }
  }
  int bdy = bi / Kn, bdx = bi % Kn;
  const _Float16* rsel = rpad + ((size_t)((y + bdy) * RP + (x + bdx))) * Ec;
  _Float16* o = minp + (size_t)idx * SEC;             // stb -> channels 0..63
  for (int c = 0; c < Ec; ++c) o[c] = rsel[c];
  coor[((size_t)s * 2 + 0) * Pn + p] = bdy - DISPn;
  coor[((size_t)s * 2 + 1) * Pn + p] = bdx - DISPn;
}

// ---------------------------------------------------------------------------
// SE: global average pool per (s,c) over m_input (f16 -> f32 accumulate).
// ---------------------------------------------------------------------------
__global__ void k_se_pool(const _Float16* __restrict__ minp, float* __restrict__ pool) {
  int c = blockIdx.x % SEC;
  int s = blockIdx.x / SEC;
  float sum = 0.f;
  for (int p = threadIdx.x; p < Pn; p += blockDim.x)
    sum += (float)minp[((size_t)s * Pn + p) * SEC + c];
  __shared__ float sh[256];
  sh[threadIdx.x] = sum;
  __syncthreads();
  for (int st = 128; st > 0; st >>= 1) {
    if (threadIdx.x < st) sh[threadIdx.x] += sh[threadIdx.x + st];
    __syncthreads();
  }
  if (threadIdx.x == 0) pool[(size_t)s * SEC + c] = sh[0] * (1.0f / Pn);
}

// SE: fc1(256->16)+ReLU, fc2(16->256)+sigmoid. One block per frame.
__global__ void k_se_fc(const float* __restrict__ pool, const float* __restrict__ fc1,
                        const float* __restrict__ fc2, float* __restrict__ gate) {
  int s = blockIdx.x, t = threadIdx.x;                // 256 threads
  __shared__ float sp[SEC];
  __shared__ float sh[SEH];
  sp[t] = pool[(size_t)s * SEC + t];
  __syncthreads();
  if (t < SEH) {
    float a = 0.f;
    for (int j = 0; j < SEC; ++j) a += fc1[t * SEC + j] * sp[j];
    sh[t] = fmaxf(a, 0.0f);
  }
  __syncthreads();
  float a = 0.f;
  for (int j = 0; j < SEH; ++j) a += fc2[t * SEH + j] * sh[j];
  gate[(size_t)s * SEC + t] = 1.0f / (1.0f + expf(-a));
}

// SE: scale m_input channel-last f16 by per-(s,c) gate.
__global__ void k_se_scale(_Float16* __restrict__ minp, const float* __restrict__ gate) {
  size_t idx = (size_t)blockIdx.x * blockDim.x + threadIdx.x;
  if (idx >= (size_t)Sn * Pn * SEC) return;
  int c = (int)(idx % SEC);
  int s = (int)(idx / ((size_t)Pn * SEC));
  minp[idx] = (_Float16)((float)minp[idx] * gate[(size_t)s * SEC + c]);
}

// ---------------------------------------------------------------------------
// Host-side orchestration
// ---------------------------------------------------------------------------
extern "C" void kernel_launch(void* const* d_in, const int* in_sizes, int n_in,
                              void* d_out, int out_size, void* d_ws, size_t ws_size,
                              hipStream_t stream) {
  (void)in_sizes; (void)n_in; (void)out_size; (void)ws_size;

  const float* feat_hi  = (const float*)d_in[0];
  const float* feat_lo  = (const float*)d_in[1];
  const float* seg      = (const float*)d_in[2];
  const float* feat_w   = (const float*)d_in[3];
  const float* feat_b   = (const float*)d_in[4];
  const float* feat_g   = (const float*)d_in[5];
  const float* feat_be  = (const float*)d_in[6];
  const float* embA_w   = (const float*)d_in[7];
  const float* embA_b   = (const float*)d_in[8];
  const float* embA_g   = (const float*)d_in[9];
  const float* embA_be  = (const float*)d_in[10];
  const float* embB_w   = (const float*)d_in[11];
  const float* embB_b   = (const float*)d_in[12];
  const float* embB_g   = (const float*)d_in[13];
  const float* embB_be  = (const float*)d_in[14];
  const float* segA_w   = (const float*)d_in[15];
  const float* segA_b   = (const float*)d_in[16];
  const float* segA_g   = (const float*)d_in[17];
  const float* segA_be  = (const float*)d_in[18];
  const float* segB_w   = (const float*)d_in[19];
  const float* segB_b   = (const float*)d_in[20];
  const float* segB_g   = (const float*)d_in[21];
  const float* segB_be  = (const float*)d_in[22];
  const float* se_fc1   = (const float*)d_in[23];
  const float* se_fc2   = (const float*)d_in[24];
  const float* merge_w  = (const float*)d_in[25];
  const float* merge_b  = (const float*)d_in[26];
  const float* merge_g  = (const float*)d_in[27];
  const float* merge_be = (const float*)d_in[28];
  const float* out_w    = (const float*)d_in[29];
  const float* out_b    = (const float*)d_in[30];

  float* m_out = (float*)d_out;                           // [S,2,H,W] fp32
  int*   coor  = (int*)(m_out + (size_t)Sn * 2 * Pn);     // [S,2,H,W] int32

  // Workspace carve-up (256B aligned)
  char* ws = (char*)d_ws;
  size_t off = 0;
  auto alloc = [&](size_t bytes) -> void* {
    void* p = ws + off;
    off = (off + bytes + 255) & ~(size_t)255;
    return p;
  };
  _Float16* w_feat  = (_Float16*)alloc((size_t)Fc * 1 * CINF * 2);
  _Float16* w_embA  = (_Float16*)alloc((size_t)Ec * 9 * Fc   * 2);
  _Float16* w_embB  = (_Float16*)alloc((size_t)Ec * 9 * Ec   * 2);
  _Float16* w_segA  = (_Float16*)alloc((size_t)Ec * 9 * 96   * 2);
  _Float16* w_segB  = (_Float16*)alloc((size_t)Ec * 9 * Ec   * 2);
  _Float16* w_merge = (_Float16*)alloc((size_t)Ec * 9 * SEC  * 2);
  _Float16* w_out   = (_Float16*)alloc((size_t)16 * 9 * Ec   * 2);
  _Float16* minpCL  = (_Float16*)alloc((size_t)Sn * Pn * SEC * 2);  // m_input [S][P][256]
  float*    tbuf    = (float*)   alloc((size_t)Sn * Fc * Pn * 4);   // NCHW conv scratch
  _Float16* miCL    = (_Float16*)alloc((size_t)Sn * Pn * Fc * 2);   // matching_input CL
  _Float16* tCL     = (_Float16*)alloc((size_t)Sn * Pn * Ec * 2);   // embA out CL
  _Float16* qCL     = (_Float16*)alloc((size_t)Sn * Pn * Ec * 2);   // q CL
  _Float16* refCL   = (_Float16*)alloc((size_t)Pn * 96 * 2);        // ref_in CL (padded)
  _Float16* rCL     = (_Float16*)alloc((size_t)Pn * Ec * 2);        // seg branch CL
  _Float16* rpadCL  = (_Float16*)alloc((size_t)RP * RP * Ec * 2);   // r_pad CL
  _Float16* mCL     = (_Float16*)alloc((size_t)Sn * Pn * Ec * 2);   // merge out CL
  float*    pool    = (float*)   alloc((size_t)Sn * SEC * 4);
  float*    gate    = (float*)   alloc((size_t)Sn * SEC * 4);

  auto reorder = [&](const float* src, _Float16* dst, int OC, int IC, int KK,
                     int OCp, int ICp) {
    int total = OCp * KK * KK * ICp;
    k_reorder_w<<<(total + 255) / 256, 256, 0, stream>>>(src, dst, OC, IC, KK, OCp, ICp);
  };
  reorder(feat_w,  w_feat,  Fc, CINF, 1, Fc, CINF);
  reorder(embA_w,  w_embA,  Ec, Fc,   3, Ec, Fc);
  reorder(embB_w,  w_embB,  Ec, Ec,   3, Ec, Ec);
  reorder(segA_w,  w_segA,  Ec, 66,   3, Ec, 96);
  reorder(segB_w,  w_segB,  Ec, Ec,   3, Ec, Ec);
  reorder(merge_w, w_merge, Ec, SEC,  3, Ec, SEC);
  reorder(out_w,   w_out,   2,  Ec,   3, 16, Ec);

  // 1) feature = cat(feat_hi, up(feat_lo)) -> m_input channels 64..255 (f16 CL)
  k_build_feature<<<(Sn * Pn + 255) / 256, 256, 0, stream>>>(feat_hi, feat_lo, minpCL);

  const dim3 gconv(Pn / 16, 1, Sn);    // (1024, 1, 8) — one wave per 16-px tile
  const dim3 gconv1(Pn / 16, 1, 1);
  const size_t sm_feat = (size_t)1 * 16 * CINF * 2;   // 1x1 strip
  const size_t sm_64   = (size_t)3 * 18 * Ec   * 2;   // 3x3, cin=64
  const size_t sm_96   = (size_t)3 * 18 * 96   * 2;   // 3x3, cin=96
  const size_t sm_256  = (size_t)3 * 18 * SEC  * 2;   // 3x3, cin=256

  // 2) matching_input = cna(feature, 1x1 192->64)
  k_conv_wmma<<<gconv, 32, sm_feat, stream>>>(minpCL, SEC, Ec, CINF, 1, w_feat,
                                              feat_b, Fc, 4, tbuf, 1);
  k_instnorm<<<Sn * Fc, 256, 0, stream>>>(tbuf, feat_g, feat_be, miCL, Fc, Fc);

  // 3) q = cna(cna(matching_input, embA), embB)  (3x3 64->64 twice)
  k_conv_wmma<<<gconv, 32, sm_64, stream>>>(miCL, Fc, 0, Fc, 3, w_embA, embA_b,
                                            Ec, 4, tbuf, 1);
  k_instnorm<<<Sn * Ec, 256, 0, stream>>>(tbuf, embA_g, embA_be, tCL, Ec, Ec);
  k_conv_wmma<<<gconv, 32, sm_64, stream>>>(tCL, Ec, 0, Ec, 3, w_embB, embB_b,
                                            Ec, 4, tbuf, 1);
  k_instnorm<<<Sn * Ec, 256, 0, stream>>>(tbuf, embB_g, embB_be, qCL, Ec, Ec);

  // 4) reference embedding r (frame 0): cat(mi[0], fg, bg) -> segA -> segB
  k_build_ref<<<(Pn + 255) / 256, 256, 0, stream>>>(miCL, seg, refCL);
  k_conv_wmma<<<gconv1, 32, sm_96, stream>>>(refCL, 96, 0, 96, 3, w_segA, segA_b,
                                             Ec, 4, tbuf, 1);
  k_instnorm<<<Ec, 256, 0, stream>>>(tbuf, segA_g, segA_be, rCL, Ec, Ec);
  k_conv_wmma<<<gconv1, 32, sm_64, stream>>>(rCL, Ec, 0, Ec, 3, w_segB, segB_b,
                                             Ec, 4, tbuf, 1);
  k_instnorm<<<Ec, 256, 0, stream>>>(tbuf, segB_g, segB_be, rCL, Ec, Ec);

  // 5) pad r and run windowed correlation + argmax gather (fills stb ch 0..63)
  k_pad_r<<<(RP * RP + 255) / 256, 256, 0, stream>>>(rCL, rpadCL);
  k_corr<<<(Sn * Pn + 255) / 256, 256, 0, stream>>>(qCL, rpadCL, minpCL, coor);

  // 6) SE gating on m_input
  k_se_pool<<<Sn * SEC, 256, 0, stream>>>(minpCL, pool);
  k_se_fc<<<Sn, 256, 0, stream>>>(pool, se_fc1, se_fc2, gate);
  {
    size_t total = (size_t)Sn * Pn * SEC;
    k_se_scale<<<(unsigned)((total + 255) / 256), 256, 0, stream>>>(minpCL, gate);
  }

  // 7) merge conv (3x3 256->64) + instnorm, then out conv (3x3 64->2)
  k_conv_wmma<<<gconv, 32, sm_256, stream>>>(minpCL, SEC, 0, SEC, 3, w_merge,
                                             merge_b, Ec, 4, tbuf, 1);
  k_instnorm<<<Sn * Ec, 256, 0, stream>>>(tbuf, merge_g, merge_be, mCL, Ec, Ec);
  k_conv_wmma<<<gconv, 32, sm_64, stream>>>(mCL, Ec, 0, Ec, 3, w_out, out_b,
                                            2, 1, m_out, 0);
}
